// SubLSTM_47047071760546
// MI455X (gfx1250) — matrix operand order repaired and checked by reading, not compiled, not run on hardware
//
#include <hip/hip_runtime.h>
#include <hip/hip_bf16.h>

// ---------------- problem constants ----------------
#define T_STEPS 1024
#define B_      64
#define H_      512
#define L_      2
#define IN_     512
#define GATE_   1536              // 3*H : i_gate, o_gate, z
#define KTOT    (IN_ + H_)        // 1024 (concatenated [x | h] GEMM K)
#define KCH     64                // K-chunk staged in LDS per iteration
#define NKC     (KTOT / KCH)      // 16 chunks
#define NCHUNK  64                // hidden units handled per block
#define NBLOCKS (H_ / NCHUNK)     // 8
#define AST     72                // padded LDS row stride (bf16): 144B, 16B-aligned

typedef __attribute__((ext_vector_type(16))) __bf16        v16bf;
typedef __attribute__((ext_vector_type(8)))  float         v8f;
typedef __attribute__((ext_vector_type(8)))  unsigned int  v8u;

// ---- gfx1250 async global->LDS path (guarded; falls back to uint4 copies) ----
#if defined(__AMDGCN__) && \
    __has_builtin(__builtin_amdgcn_global_load_async_to_lds_b128) && \
    __has_builtin(__builtin_amdgcn_s_wait_asynccnt)
#define USE_ASYNC_LDS 1
typedef __attribute__((__vector_size__(16))) int v4i_;
typedef __attribute__((address_space(1))) v4i_ g_v4i;   // global int4
typedef __attribute__((address_space(3))) v4i_ l_v4i;   // LDS int4
#define COPY16(ldst, gsrc) \
    __builtin_amdgcn_global_load_async_to_lds_b128((g_v4i*)(gsrc), (l_v4i*)(ldst), 0, 0)
#else
#define USE_ASYNC_LDS 0
#define COPY16(ldst, gsrc) do { *(uint4*)(ldst) = *(const uint4*)(gsrc); } while (0)
#endif

__device__ __forceinline__ void wait_async_upto(int n) {
#if USE_ASYNC_LDS
    if (n >= 8)      __builtin_amdgcn_s_wait_asynccnt(8);
    else if (n >= 6) __builtin_amdgcn_s_wait_asynccnt(6);
    else             __builtin_amdgcn_s_wait_asynccnt(0);
#else
    (void)n;
#endif
}

__device__ __forceinline__ unsigned short f2bf(float x) {
    unsigned int u = __builtin_bit_cast(unsigned int, x);
    u += 0x7FFFu + ((u >> 16) & 1u);          // round-to-nearest-even
    return (unsigned short)(u >> 16);
}
__device__ __forceinline__ unsigned int pk2bf(float a, float b) {
    return (unsigned int)f2bf(a) | ((unsigned int)f2bf(b) << 16);
}
__device__ __forceinline__ float sigmoidf_(float x) {
    return 1.0f / (1.0f + __expf(-x));
}

// ---------------- prep: pack weights/biases/state ----------------
__global__ void sublstm_prep(const float* __restrict__ W,   // [L,GATE,IN]
                             const float* __restrict__ R,   // [L,GATE,H]
                             const float* __restrict__ bi,  // [L,GATE]
                             const float* __restrict__ bh,  // [L,GATE]
                             const float* __restrict__ f,   // [L,H]
                             const float* __restrict__ h0,  // [L,B,H]
                             const float* __restrict__ c0,  // [L,B,H]
                             unsigned short* __restrict__ WRcat, // [L,GATE,KTOT] bf16
                             float* __restrict__ bias,           // [L,GATE]
                             float* __restrict__ fgv,            // [L,H]
                             unsigned short* __restrict__ hbf,   // [2,L,B,H] bf16
                             float* __restrict__ hf32,           // [L,B,H]
                             float* __restrict__ cst)            // [L,B,H]
{
    const int tid    = blockIdx.x * blockDim.x + threadIdx.x;
    const int stride = gridDim.x * blockDim.x;

    for (int i = tid; i < L_ * GATE_ * KTOT; i += stride) {
        int k = i % KTOT;
        int g = (i / KTOT) % GATE_;
        int l = i / (KTOT * GATE_);
        float v = (k < IN_) ? W[(size_t)(l * GATE_ + g) * IN_ + k]
                            : R[(size_t)(l * GATE_ + g) * H_ + (k - IN_)];
        WRcat[i] = f2bf(v);
    }
    for (int i = tid; i < L_ * GATE_; i += stride) bias[i] = bi[i] + bh[i];
    for (int i = tid; i < L_ * H_;    i += stride) fgv[i]  = sigmoidf_(f[i]);
    for (int i = tid; i < L_ * B_ * H_; i += stride) {
        float h = h0[i];
        unsigned short hb = f2bf(h);
        hbf[i]                    = hb;   // parity 0
        hbf[L_ * B_ * H_ + i]     = hb;   // parity 1 (read at t=0)
        hf32[i] = h;
        cst[i]  = c0[i];
    }
}

// ---------------- stage one K-chunk into an LDS buffer ----------------
__device__ __forceinline__ void stage_chunk(
    int kc, unsigned short* __restrict__ Abuf, unsigned short* __restrict__ Bbuf,
    const float* __restrict__ xin,
    const unsigned short* __restrict__ hin_x,
    const unsigned short* __restrict__ hin_prev,
    const unsigned short* __restrict__ Wl,
    int l, int cb, int tid)
{
    const int kbase = kc * KCH;

    // ---- A tile: [64 rows x 64 K]
    if (kbase < IN_ && l == 0) {
        // fp32 -> bf16 convert from x (synchronous, float4 vectorized)
        int row  = tid >> 2;
        int kseg = (tid & 3) * 16;
        const float4* src = (const float4*)(xin + row * IN_ + kbase + kseg);
        float4 f0 = src[0], f1 = src[1], f2 = src[2], f3 = src[3];
        uint4 u0, u1;
        u0.x = pk2bf(f0.x, f0.y);  u0.y = pk2bf(f0.z, f0.w);
        u0.z = pk2bf(f1.x, f1.y);  u0.w = pk2bf(f1.z, f1.w);
        u1.x = pk2bf(f2.x, f2.y);  u1.y = pk2bf(f2.z, f2.w);
        u1.z = pk2bf(f3.x, f3.y);  u1.w = pk2bf(f3.z, f3.w);
        *(uint4*)&Abuf[row * AST + kseg]     = u0;
        *(uint4*)&Abuf[row * AST + kseg + 8] = u1;
    } else {
        const unsigned short* hsrc =
            (kbase < IN_) ? (hin_x + kbase) : (hin_prev + (kbase - IN_));
        #pragma unroll
        for (int o = tid; o < (B_ * KCH) / 8; o += 256) {   // 512 x b128, 2/thread
            int row = o >> 3;
            int kk  = (o & 7) * 8;
            COPY16(&Abuf[row * AST + kk], hsrc + row * H_ + kk);
        }
    }

    // ---- B tile: 192 gate cols (i,o,z thirds of this chunk) x 64 K
    #pragma unroll
    for (int o = tid; o < (3 * NCHUNK * KCH) / 8; o += 256) { // 1536 x b128, 6/thread
        int gl    = o >> 3;
        int kk    = (o & 7) * 8;
        int gglob = (gl >> 6) * H_ + cb * NCHUNK + (gl & 63);
        COPY16(&Bbuf[gl * AST + kk], Wl + (size_t)gglob * KTOT + kbase + kk);
    }
}

__device__ __forceinline__ int chunk_async_cnt(int kc, int l) {
    // async b128 ops issued per thread for chunk kc (uniform across wave)
    return 6 + (((kc * KCH) < IN_ && l == 0) ? 0 : 2);
}

// ---------------- recurrent step: one (t, layer) ----------------
__global__ __launch_bounds__(256)
void sublstm_step(const float* __restrict__ x,            // [T,B,IN]
                  const unsigned short* __restrict__ WRcat,
                  const float* __restrict__ bias,
                  const float* __restrict__ fgv,
                  unsigned short* __restrict__ hbf,       // [2,L,B,H] bf16
                  float* __restrict__ hf32,               // [L,B,H]
                  float* __restrict__ cst,                // [L,B,H]
                  float* __restrict__ out,                // [T,B,H]
                  int t, int l)
{
    __shared__ unsigned short Asl[2][B_ * AST];           // 2 x  9.0 KB
    __shared__ unsigned short Bsl[2][3 * NCHUNK * AST];   // 2 x 27.0 KB

    const int tid  = threadIdx.x;
    const int lane = tid & 31;
    const int w    = tid >> 5;       // wave 0..7
    const int cb   = blockIdx.x;     // hidden 64-chunk 0..7
    const int p    = t & 1;          // h ping-pong parity
    const int j    = w & 3;          // 16-col subtile within chunk
    const int mp   = w >> 2;         // m-tile pair {2mp, 2mp+1}

    const unsigned short* hin_prev = hbf + ((size_t)(1 - p) * L_ + l) * B_ * H_; // h_{t-1}[l]
    const unsigned short* hin_x    = hbf + ((size_t)p * L_ + 0) * B_ * H_;       // h_t[0]
    const float*          xin      = x + (size_t)t * B_ * IN_;
    const unsigned short* Wl       = WRcat + (size_t)l * GATE_ * KTOT;

    v8f acc[2][3] = {};   // [m-tile][gate: i,o,z] fp32 accumulators

    const int half = lane >> 4;
    const int lm   = lane & 15;

    // prologue: fill buffer 0
    stage_chunk(0, Asl[0], Bsl[0], xin, hin_x, hin_prev, Wl, l, cb, tid);

    for (int kc = 0; kc < NKC; ++kc) {
        // issue next chunk into the other buffer (overlaps with this chunk's WMMAs)
        if (kc + 1 < NKC)
            stage_chunk(kc + 1, Asl[(kc + 1) & 1], Bsl[(kc + 1) & 1],
                        xin, hin_x, hin_prev, Wl, l, cb, tid);
        // allow only the just-issued chunk's async ops to remain outstanding
        wait_async_upto((kc + 1 < NKC) ? chunk_async_cnt(kc + 1, l) : 0);
        __syncthreads();   // all waves' chunk-kc data resident in LDS

        const unsigned short* Ab = Asl[kc & 1];
        const unsigned short* Bb = Bsl[kc & 1];

        #pragma unroll
        for (int ks = 0; ks < KCH / 32; ++ks) {
            // A fragments (16-bit A 16x32 layout)
            v16bf av[2];
            #pragma unroll
            for (int mi = 0; mi < 2; ++mi) {
                int row = (mp * 2 + mi) * 16 + lm;
                uint4 lo = *(const uint4*)&Ab[row * AST + ks * 32 + half * 8];
                uint4 hi = *(const uint4*)&Ab[row * AST + ks * 32 + 16 + half * 8];
                v8u af = { lo.x, lo.y, lo.z, lo.w, hi.x, hi.y, hi.z, hi.w };
                av[mi] = __builtin_bit_cast(v16bf, af);
            }
            #pragma unroll
            for (int g = 0; g < 3; ++g) {
                // B fragment (32x16: lanes 0-15 K=0..7,16..23, lanes 16-31 K=8..15,24..31)
                int gl = g * NCHUNK + j * 16 + lm;
                uint4 b0 = *(const uint4*)&Bb[gl * AST + ks * 32 + half * 8];
                uint4 b1 = *(const uint4*)&Bb[gl * AST + ks * 32 + 16 + half * 8];
                v8u bf = { b0.x, b0.y, b0.z, b0.w, b1.x, b1.y, b1.z, b1.w };
                v16bf bv = __builtin_bit_cast(v16bf, bf);
                #pragma unroll
                for (int mi = 0; mi < 2; ++mi) {
                    acc[mi][g] = __builtin_amdgcn_wmma_f32_16x16x32_bf16(
                        false, av[mi], false, bv, (short)0, acc[mi][g], false, false);
                }
            }
        }
        __syncthreads();   // everyone done reading buf[kc&1] before it is overwritten
    }

    // ---- pointwise gate update (each wave owns matched i/o/z columns)
    const int   ncol = cb * NCHUNK + j * 16 + lm;       // hidden unit index
    const float bi_i = bias[l * GATE_ + ncol];
    const float bi_o = bias[l * GATE_ + H_ + ncol];
    const float bi_z = bias[l * GATE_ + 2 * H_ + ncol];
    const float fg   = fgv[l * H_ + ncol];

    unsigned short* hout = hbf + ((size_t)p * L_ + l) * B_ * H_;
    float* hf = hf32 + (size_t)l * B_ * H_;
    float* cl = cst  + (size_t)l * B_ * H_;

    #pragma unroll
    for (int mi = 0; mi < 2; ++mi) {
        #pragma unroll
        for (int r = 0; r < 8; ++r) {
            int row = (mp * 2 + mi) * 16 + half * 8 + r;   // C/D layout: VGPR r, lane halves
            float ig = sigmoidf_(acc[mi][0][r] + bi_i);
            float og = sigmoidf_(acc[mi][1][r] + bi_o);
            float zg = sigmoidf_(acc[mi][2][r] + bi_z);
            size_t idx = (size_t)row * H_ + ncol;
            float cn = cl[idx] * fg + zg - ig;
            float hn = sigmoidf_(cn) - og;
            cl[idx]   = cn;
            hf[idx]   = hn;
            hout[idx] = f2bf(hn);
            if (l == 1) out[(size_t)t * B_ * H_ + idx] = hn;
        }
    }
}

// ---------------- host side ----------------
extern "C" void kernel_launch(void* const* d_in, const int* in_sizes, int n_in,
                              void* d_out, int out_size, void* d_ws, size_t ws_size,
                              hipStream_t stream) {
    (void)in_sizes; (void)n_in; (void)out_size; (void)ws_size;
    const float* x  = (const float*)d_in[0];
    const float* h0 = (const float*)d_in[1];
    const float* c0 = (const float*)d_in[2];
    const float* W  = (const float*)d_in[3];
    const float* R  = (const float*)d_in[4];
    const float* bi = (const float*)d_in[5];
    const float* bh = (const float*)d_in[6];
    const float* f  = (const float*)d_in[7];

    char* ws = (char*)d_ws;
    size_t off = 0;
    unsigned short* WRcat = (unsigned short*)(ws + off); off += (size_t)L_ * GATE_ * KTOT * 2;
    float*          bias  = (float*)(ws + off);          off += (size_t)L_ * GATE_ * 4;
    float*          fgv   = (float*)(ws + off);          off += (size_t)L_ * H_ * 4;
    unsigned short* hbf   = (unsigned short*)(ws + off); off += (size_t)2 * L_ * B_ * H_ * 2;
    float*          hf32  = (float*)(ws + off);          off += (size_t)L_ * B_ * H_ * 4;
    float*          cst   = (float*)(ws + off);          off += (size_t)L_ * B_ * H_ * 4;

    float* out = (float*)d_out;

    sublstm_prep<<<512, 256, 0, stream>>>(W, R, bi, bh, f, h0, c0,
                                          WRcat, bias, fgv, hbf, hf32, cst);

    for (int t = 0; t < T_STEPS; ++t) {
        sublstm_step<<<NBLOCKS, 256, 0, stream>>>(x, WRcat, bias, fgv, hbf, hf32, cst, out, t, 0);
        sublstm_step<<<NBLOCKS, 256, 0, stream>>>(x, WRcat, bias, fgv, hbf, hf32, cst, out, t, 1);
    }

    // tuple tail: h_f [L,B,H] then c_f [L,B,H]
    float* tail = out + (size_t)T_STEPS * B_ * H_;
    (void)hipMemcpyAsync(tail,                hf32, (size_t)L_ * B_ * H_ * sizeof(float),
                         hipMemcpyDeviceToDevice, stream);
    (void)hipMemcpyAsync(tail + L_ * B_ * H_, cst,  (size_t)L_ * B_ * H_ * sizeof(float),
                         hipMemcpyDeviceToDevice, stream);
}
